// TFSF_TF_15582141350533
// MI455X (gfx1250) — matrix operand order repaired and checked by reading, not compiled, not run on hardware
//
#include <hip/hip_runtime.h>
#include <math.h>

// ---------------- problem constants ----------------
#define NNODES 50000
#define TSTEPS 5
#define NEDGE  1600000
#define FRAW   395
#define HGRU   64
#define HID    40
#define NROWS  (NNODES * TSTEPS)        // 250000, divisible by 16
#define ROWTILES (NROWS / 16)           // 15625
#define NODETILES (NNODES / 16)         // 3125

typedef __attribute__((ext_vector_type(16))) __bf16 v16bf;
typedef __attribute__((ext_vector_type(8)))  float  v8f;

static __device__ __forceinline__ v8f wmma_bf16(v16bf a, v16bf b, v8f c) {
    // D = A(16x32 bf16) * B(32x16 bf16) + C(16x16 f32)
    return __builtin_amdgcn_wmma_f32_16x16x32_bf16(
        /*neg_a=*/false, a, /*neg_b=*/false, b,
        /*c_mod=*/(short)0, c, /*reuse_a=*/false, /*reuse_b=*/false);
}

// A-matrix fragment (16x32 bf16) from row-major buffer (LDS or global).
// ISA layout: lane-lo holds K 0..7 (vgpr0-3) and 16..23 (vgpr4-7) of row m=lane&15;
// lane-hi holds K 8..15 and 24..31.
static __device__ __forceinline__ v16bf load_A(const __bf16* A, int ld, int lane) {
    int m  = lane & 15;
    int kb = (lane >> 4) * 8;
    const __bf16* p = A + m * ld + kb;
    v16bf a;
#pragma unroll
    for (int j = 0; j < 8; ++j) { a[j] = p[j]; a[8 + j] = p[16 + j]; }
    return a;
}

// B fragment from a pre-packed fragment image: 32 lanes x 16 bf16 contiguous.
static __device__ __forceinline__ v16bf load_pack(const __bf16* frag, int lane) {
    return *(const v16bf*)(frag + lane * 16);
}

static __device__ __forceinline__ float sigmoidf_(float x) {
    return 1.0f / (1.0f + __expf(-x));
}

// ---------------- pack kernel: build B-fragment images ----------------
// B(32x16) VGPR layout: lane n = lane&15 is the output column; lane-lo covers
// K 0..7,16..23 (kb=0), lane-hi K 8..15,24..31 (kb=8). Element e = B[kb+(e&7)+16*(e>>3)][n].
// wtrans=1: W stored [Nout, K] (use W[jg*K+ig])   -> B = W^T  (x @ W^T)
// wtrans=0: W stored [K, Nout] (use W[ig*Nout+jg]) -> B = W    (x @ W)
__global__ void pack_bfrag_kernel(const float* __restrict__ W, __bf16* __restrict__ pack,
                                  int K, int Nout, int nkk, int nnt, int wtrans) {
    int total = nkk * nnt * 512;
    for (int f = blockIdx.x * blockDim.x + threadIdx.x; f < total;
         f += gridDim.x * blockDim.x) {
        int frag = f >> 9, r = f & 511;
        int lane = r >> 4, e = r & 15;
        int kk = frag / nnt, nt = frag % nnt;
        int n  = lane & 15;
        int kb = (lane >> 4) * 8;
        int ig = kk * 32 + kb + (e & 7) + ((e >> 3) << 4);      // K index
        int jg = nt * 16 + n;           // output column
        float v = 0.0f;
        if (ig < K && jg < Nout)
            v = wtrans ? W[jg * K + ig] : W[ig * Nout + jg];
        pack[f] = (__bf16)v;
    }
}

// ---------------- degree / normalization ----------------
__global__ void deg_init_kernel(unsigned* __restrict__ d1, unsigned* __restrict__ d2) {
    int i = blockIdx.x * blockDim.x + threadIdx.x;
    if (i < NNODES) { d1[i] = 1u; d2[i] = 1u; }   // self-loop
}
__global__ void deg_count_kernel(const int* __restrict__ ei1, const int* __restrict__ ei2,
                                 unsigned* __restrict__ d1, unsigned* __restrict__ d2) {
    int e = blockIdx.x * blockDim.x + threadIdx.x;
    if (e < NEDGE) {
        atomicAdd(&d1[ei1[NEDGE + e]], 1u);
        atomicAdd(&d2[ei2[NEDGE + e]], 1u);
    }
}
__global__ void dinv_kernel(const unsigned* __restrict__ d1, const unsigned* __restrict__ d2,
                            float* __restrict__ v1, float* __restrict__ v2) {
    int i = blockIdx.x * blockDim.x + threadIdx.x;
    if (i < NNODES) {
        v1[i] = rsqrtf((float)d1[i]);
        v2[i] = rsqrtf((float)d2[i]);
    }
}

// ---------------- kernel 1: Conv1d + Linear (per row), WMMA for the 80->40 ----
#define K1_WAVES 8
__global__ void cnn_lin_kernel(const float* __restrict__ x,
                               const float* __restrict__ conv_w,
                               const float* __restrict__ conv_b,
                               const __bf16* __restrict__ lin1_pack,
                               const float* __restrict__ lin1_b,
                               float* __restrict__ seq) {
    __shared__ __bf16 Cs[K1_WAVES][16][96];     // conv outputs, K padded 80->96
    int wave = threadIdx.x >> 5, lane = threadIdx.x & 31;
    int tile = blockIdx.x * K1_WAVES + wave;
    bool active = tile < ROWTILES;
    int row0 = active ? tile * 16 : 0;

    // zero K-padding columns 80..95
    for (int i = lane; i < 16 * 16; i += 32)
        Cs[wave][i >> 4][80 + (i & 15)] = (__bf16)0.0f;

    // conv1d(1->16, k=40, s=88) + relu, 16 rows x 80 outputs per wave
    for (int it = 0; it < 40; ++it) {
        int e = it * 32 + lane;          // 0..1279
        int m = e / 80, idx = e - m * 80;
        int oc = idx / 5, p = idx - oc * 5;
        const float* sig = x + (size_t)(row0 + m) * FRAW + 3 + p * 88;
        const float* w   = conv_w + oc * 40;
        float acc = conv_b[oc];
#pragma unroll 8
        for (int k = 0; k < 40; ++k) acc = fmaf(sig[k], w[k], acc);
        Cs[wave][m][idx] = (__bf16)fmaxf(acc, 0.0f);
    }
    __syncthreads();

    if (active) {
        v16bf afr[3];
#pragma unroll
        for (int kk = 0; kk < 3; ++kk)
            afr[kk] = load_A(&Cs[wave][0][0] + kk * 32, 96, lane);
        int n = lane & 15, hi = lane >> 4;
#pragma unroll
        for (int nt = 0; nt < 3; ++nt) {
            v8f acc = {};
#pragma unroll
            for (int kk = 0; kk < 3; ++kk)
                acc = wmma_bf16(afr[kk], load_pack(lin1_pack + (kk * 3 + nt) * 512, lane), acc);
            int col = nt * 16 + n;
            if (col < 40) {
                float bc = lin1_b[col];
#pragma unroll
                for (int v = 0; v < 8; ++v) {
                    int row = row0 + v + 8 * hi;
                    seq[(size_t)row * 41 + col] = fmaxf(acc[v] + bc, 0.0f);
                }
            }
        }
        if (lane < 16) {  // y channel
            int row = row0 + lane;
            seq[(size_t)row * 41 + 40] = x[(size_t)row * FRAW + 2];
        }
    }
}

// ---------------- kernel 2: GRU over t=0..3, WMMA gates -------------------
#define K2_WAVES 8
__global__ void gru_kernel(const float* __restrict__ seq,
                           const __bf16* __restrict__ wih_pack,  // K=41->64, N=192
                           const __bf16* __restrict__ whh_pack,  // K=64,    N=192
                           const float* __restrict__ b_ih,
                           const float* __restrict__ b_hh,
                           __bf16* __restrict__ feats) {          // [N][128] padded
    __shared__ __bf16 Ax[K2_WAVES][16][64];
    __shared__ __bf16 Hb[K2_WAVES][16][64];
    int wave = threadIdx.x >> 5, lane = threadIdx.x & 31;
    int tile = blockIdx.x * K2_WAVES + wave;
    bool active = tile < NODETILES;
    int node0 = active ? tile * 16 : 0;
    int n = lane & 15, hi = lane >> 4;

    float h[4][8];   // h[cg][v] : lane owns (m = v+8*hi, c = cg*16+n)
#pragma unroll
    for (int cg = 0; cg < 4; ++cg)
#pragma unroll
        for (int v = 0; v < 8; ++v) h[cg][v] = 0.0f;

    for (int i = lane; i < 16 * 64; i += 32)
        (&Hb[wave][0][0])[i] = (__bf16)0.0f;

    for (int t = 0; t < 4; ++t) {
        // stage x_t (41 cols, padded to 64)
        for (int i = lane; i < 16 * 64; i += 32) {
            int m = i >> 6, c = i & 63;
            float v = (c < 41) ? seq[((size_t)(node0 + m) * 5 + t) * 41 + c] : 0.0f;
            Ax[wave][m][c] = (__bf16)v;
        }
        __syncthreads();
        v16bf ax0 = load_A(&Ax[wave][0][0],      64, lane);
        v16bf ax1 = load_A(&Ax[wave][0][0] + 32, 64, lane);
        v16bf ah0 = load_A(&Hb[wave][0][0],      64, lane);
        v16bf ah1 = load_A(&Hb[wave][0][0] + 32, 64, lane);
        __syncthreads();   // fragments read before Hb is overwritten below

#pragma unroll
        for (int cg = 0; cg < 4; ++cg) {
            v8f ir = {}, iz = {}, in_ = {}, hr = {}, hz = {}, hn = {};
            // gi = x @ W_ih^T : gate tiles nt = cg (r), 4+cg (z), 8+cg (n)
            ir  = wmma_bf16(ax0, load_pack(wih_pack + (0 * 12 + cg)     * 512, lane), ir);
            ir  = wmma_bf16(ax1, load_pack(wih_pack + (1 * 12 + cg)     * 512, lane), ir);
            iz  = wmma_bf16(ax0, load_pack(wih_pack + (0 * 12 + 4 + cg) * 512, lane), iz);
            iz  = wmma_bf16(ax1, load_pack(wih_pack + (1 * 12 + 4 + cg) * 512, lane), iz);
            in_ = wmma_bf16(ax0, load_pack(wih_pack + (0 * 12 + 8 + cg) * 512, lane), in_);
            in_ = wmma_bf16(ax1, load_pack(wih_pack + (1 * 12 + 8 + cg) * 512, lane), in_);
            // gh = h @ W_hh^T
            hr  = wmma_bf16(ah0, load_pack(whh_pack + (0 * 12 + cg)     * 512, lane), hr);
            hr  = wmma_bf16(ah1, load_pack(whh_pack + (1 * 12 + cg)     * 512, lane), hr);
            hz  = wmma_bf16(ah0, load_pack(whh_pack + (0 * 12 + 4 + cg) * 512, lane), hz);
            hz  = wmma_bf16(ah1, load_pack(whh_pack + (1 * 12 + 4 + cg) * 512, lane), hz);
            hn  = wmma_bf16(ah0, load_pack(whh_pack + (0 * 12 + 8 + cg) * 512, lane), hn);
            hn  = wmma_bf16(ah1, load_pack(whh_pack + (1 * 12 + 8 + cg) * 512, lane), hn);

            int c = cg * 16 + n;
            float bir = b_ih[c], biz = b_ih[64 + c], bin = b_ih[128 + c];
            float bhr = b_hh[c], bhz = b_hh[64 + c], bhn = b_hh[128 + c];
#pragma unroll
            for (int v = 0; v < 8; ++v) {
                float r  = sigmoidf_(ir[v] + bir + hr[v] + bhr);
                float z  = sigmoidf_(iz[v] + biz + hz[v] + bhz);
                float ng = tanhf(in_[v] + bin + r * (hn[v] + bhn));
                h[cg][v] = (1.0f - z) * ng + z * h[cg][v];
            }
        }
        // publish h as bf16 for next step's A operand
#pragma unroll
        for (int cg = 0; cg < 4; ++cg) {
            int c = cg * 16 + n;
#pragma unroll
            for (int v = 0; v < 8; ++v)
                Hb[wave][v + 8 * hi][c] = (__bf16)h[cg][v];
        }
    }

    if (active) {
        // feats[:, 40..103] = h_n
#pragma unroll
        for (int cg = 0; cg < 4; ++cg) {
            int c = cg * 16 + n;
#pragma unroll
            for (int v = 0; v < 8; ++v)
                feats[(size_t)(node0 + v + 8 * hi) * 128 + 40 + c] = (__bf16)h[cg][v];
        }
        // feats[:, 0..39] = seq[:,4,:40] ; cols 104..127 zero-pad
        for (int i = lane; i < 16 * 128; i += 32) {
            int m = i >> 7, c = i & 127;
            if (c < 40)
                feats[(size_t)(node0 + m) * 128 + c] =
                    (__bf16)seq[((size_t)(node0 + m) * 5 + 4) * 41 + c];
            else if (c >= 104)
                feats[(size_t)(node0 + m) * 128 + c] = (__bf16)0.0f;
        }
    }
}

// ---------------- kernel 3: feats @ gcn_w (both graphs), WMMA -------------
#define K3_WAVES 8
__global__ void gcn_xw_kernel(const __bf16* __restrict__ feats,
                              const __bf16* __restrict__ g1_pack,
                              const __bf16* __restrict__ g2_pack,
                              float* __restrict__ hw1, float* __restrict__ hw2) {
    int wave = threadIdx.x >> 5, lane = threadIdx.x & 31;
    int tile = blockIdx.x * K3_WAVES + wave;
    if (tile >= NODETILES) return;   // no barriers in this kernel
    int node0 = tile * 16;
    int n = lane & 15, hi = lane >> 4;

    v16bf af[4];
#pragma unroll
    for (int kk = 0; kk < 4; ++kk)
        af[kk] = load_A(feats + (size_t)node0 * 128 + kk * 32, 128, lane);

    const __bf16* packs[2] = { g1_pack, g2_pack };
    float* outs[2] = { hw1, hw2 };
#pragma unroll
    for (int g = 0; g < 2; ++g) {
#pragma unroll
        for (int nt = 0; nt < 3; ++nt) {
            v8f acc = {};
#pragma unroll
            for (int kk = 0; kk < 4; ++kk)
                acc = wmma_bf16(af[kk], load_pack(packs[g] + (kk * 3 + nt) * 512, lane), acc);
            int col = nt * 16 + n;
            if (col < HID) {
#pragma unroll
                for (int v = 0; v < 8; ++v)
                    outs[g][(size_t)(node0 + v + 8 * hi) * HID + col] = acc[v];
            }
        }
    }
}

// ---------------- aggregation: self-loop init + edge scatter --------------
__global__ void agg_init_kernel(const float* __restrict__ hw1, const float* __restrict__ dv1,
                                const float* __restrict__ hw2, const float* __restrict__ dv2,
                                float* __restrict__ a1, float* __restrict__ a2) {
    int t = blockIdx.x * blockDim.x + threadIdx.x;
    if (t < NNODES * HID) {
        int i = t / HID;
        float w1 = dv1[i] * dv1[i], w2 = dv2[i] * dv2[i];
        a1[t] = hw1[t] * w1;
        a2[t] = hw2[t] * w2;
    }
}

__global__ void scatter_kernel(const int* __restrict__ ei, const float* __restrict__ hw,
                               const float* __restrict__ dinv, float* __restrict__ agg) {
    int t = blockIdx.x * blockDim.x + threadIdx.x;
    if (t >= NEDGE * 10) return;
    int e = t / 10, q = t - e * 10;          // 10 quads of 4 channels
    int s = ei[e], d = ei[NEDGE + e];
    float w = dinv[s] * dinv[d];
    const float4 v = *(const float4*)(hw + (size_t)s * HID + q * 4);
    float* base = agg + (size_t)d * HID + q * 4;
    unsafeAtomicAdd(base + 0, v.x * w);
    unsafeAtomicAdd(base + 1, v.y * w);
    unsafeAtomicAdd(base + 2, v.z * w);
    unsafeAtomicAdd(base + 3, v.w * w);
}

// ---------------- final: bias + relu + MLP(80->1) -------------------------
__global__ void final_kernel(const float* __restrict__ a1, const float* __restrict__ a2,
                             const float* __restrict__ b1, const float* __restrict__ b2,
                             const float* __restrict__ mlp_w, const float* __restrict__ mlp_b,
                             float* __restrict__ out) {
    int i = blockIdx.x * blockDim.x + threadIdx.x;
    if (i >= NNODES) return;
    float acc = mlp_b[0];
#pragma unroll 8
    for (int c = 0; c < HID; ++c)
        acc = fmaf(fmaxf(a1[(size_t)i * HID + c] + b1[c], 0.0f), mlp_w[c], acc);
#pragma unroll 8
    for (int c = 0; c < HID; ++c)
        acc = fmaf(fmaxf(a2[(size_t)i * HID + c] + b2[c], 0.0f), mlp_w[HID + c], acc);
    out[i] = acc;
}

// ---------------- launch ---------------------------------------------------
static inline size_t alignup(size_t x) { return (x + 255) & ~(size_t)255; }

extern "C" void kernel_launch(void* const* d_in, const int* in_sizes, int n_in,
                              void* d_out, int out_size, void* d_ws, size_t ws_size,
                              hipStream_t stream) {
    const float* x        = (const float*)d_in[0];
    const int*   ei1      = (const int*)d_in[1];
    const int*   ei2      = (const int*)d_in[2];
    const float* conv_w   = (const float*)d_in[3];
    const float* conv_b   = (const float*)d_in[4];
    const float* lin1_w   = (const float*)d_in[5];
    const float* lin1_b   = (const float*)d_in[6];
    const float* gru_w_ih = (const float*)d_in[7];
    const float* gru_w_hh = (const float*)d_in[8];
    const float* gru_b_ih = (const float*)d_in[9];
    const float* gru_b_hh = (const float*)d_in[10];
    const float* gcn1_w   = (const float*)d_in[11];
    const float* gcn1_b   = (const float*)d_in[12];
    const float* gcn2_w   = (const float*)d_in[13];
    const float* gcn2_b   = (const float*)d_in[14];
    const float* mlp_w    = (const float*)d_in[15];
    const float* mlp_b    = (const float*)d_in[16];
    float* out = (float*)d_out;

    // workspace carve-up
    char* p = (char*)d_ws;
    float*    seq   = (float*)p;    p += alignup((size_t)NROWS * 41 * 4);
    __bf16*   feats = (__bf16*)p;   p += alignup((size_t)NNODES * 128 * 2);
    float*    hw1   = (float*)p;    p += alignup((size_t)NNODES * HID * 4);
    float*    hw2   = (float*)p;    p += alignup((size_t)NNODES * HID * 4);
    float*    agg1  = (float*)p;    p += alignup((size_t)NNODES * HID * 4);
    float*    agg2  = (float*)p;    p += alignup((size_t)NNODES * HID * 4);
    unsigned* deg1  = (unsigned*)p; p += alignup((size_t)NNODES * 4);
    unsigned* deg2  = (unsigned*)p; p += alignup((size_t)NNODES * 4);
    float*    dinv1 = (float*)p;    p += alignup((size_t)NNODES * 4);
    float*    dinv2 = (float*)p;    p += alignup((size_t)NNODES * 4);
    __bf16* pk_lin1 = (__bf16*)p;   p += alignup((size_t)3 * 3 * 512 * 2);
    __bf16* pk_wih  = (__bf16*)p;   p += alignup((size_t)2 * 12 * 512 * 2);
    __bf16* pk_whh  = (__bf16*)p;   p += alignup((size_t)2 * 12 * 512 * 2);
    __bf16* pk_g1   = (__bf16*)p;   p += alignup((size_t)4 * 3 * 512 * 2);
    __bf16* pk_g2   = (__bf16*)p;   p += alignup((size_t)4 * 3 * 512 * 2);

    // 0) pack all weight matrices into WMMA B-fragment images (bf16)
    pack_bfrag_kernel<<<18, 256, 0, stream>>>(lin1_w,   pk_lin1, 80,  40,  3, 3,  1);
    pack_bfrag_kernel<<<48, 256, 0, stream>>>(gru_w_ih, pk_wih,  41,  192, 2, 12, 1);
    pack_bfrag_kernel<<<48, 256, 0, stream>>>(gru_w_hh, pk_whh,  64,  192, 2, 12, 1);
    pack_bfrag_kernel<<<24, 256, 0, stream>>>(gcn1_w,   pk_g1,   104, 40,  4, 3,  0);
    pack_bfrag_kernel<<<24, 256, 0, stream>>>(gcn2_w,   pk_g2,   104, 40,  4, 3,  0);

    // 1) degrees + symmetric normalization
    deg_init_kernel<<<(NNODES + 255) / 256, 256, 0, stream>>>(deg1, deg2);
    deg_count_kernel<<<(NEDGE + 255) / 256, 256, 0, stream>>>(ei1, ei2, deg1, deg2);
    dinv_kernel<<<(NNODES + 255) / 256, 256, 0, stream>>>(deg1, deg2, dinv1, dinv2);

    // 2) Conv1d + Linear per (node, t) row -> seq[N*5][41]
    cnn_lin_kernel<<<(ROWTILES + K1_WAVES - 1) / K1_WAVES, 32 * K1_WAVES, 0, stream>>>(
        x, conv_w, conv_b, pk_lin1, lin1_b, seq);

    // 3) GRU over 4 timesteps -> feats[N][128] (bf16, zero-padded)
    gru_kernel<<<(NODETILES + K2_WAVES - 1) / K2_WAVES, 32 * K2_WAVES, 0, stream>>>(
        seq, pk_wih, pk_whh, gru_b_ih, gru_b_hh, feats);

    // 4) feats @ W for both GCNs
    gcn_xw_kernel<<<(NODETILES + K3_WAVES - 1) / K3_WAVES, 32 * K3_WAVES, 0, stream>>>(
        feats, pk_g1, pk_g2, hw1, hw2);

    // 5) aggregation: self-loops then edge scatter
    agg_init_kernel<<<(NNODES * HID + 255) / 256, 256, 0, stream>>>(
        hw1, dinv1, hw2, dinv2, agg1, agg2);
    scatter_kernel<<<(NEDGE * 10 + 255) / 256, 256, 0, stream>>>(ei1, hw1, dinv1, agg1);
    scatter_kernel<<<(NEDGE * 10 + 255) / 256, 256, 0, stream>>>(ei2, hw2, dinv2, agg2);

    // 6) bias + relu + MLP
    final_kernel<<<(NNODES + 255) / 256, 256, 0, stream>>>(
        agg1, agg2, gcn1_b, gcn2_b, mlp_w, mlp_b, out);
}
